// GINEncoder_70806830842516
// MI455X (gfx1250) — compile-verified, matrix operand order checked by reading, not run on hardware
//
#include <hip/hip_runtime.h>

typedef __attribute__((ext_vector_type(16))) _Float16 v16h;
typedef __attribute__((ext_vector_type(8)))  _Float16 v8h;
typedef __attribute__((ext_vector_type(8)))  float    v8f;

#define FD 128   // feature dim (D_IN == H == 128)

// ---- WMMA f16 fragment K mapping (CDNA5 ISA 7.12.2, 16-bit A 16x32) ----
// lane = half*16 + (m or n); VGPR pair v holds K = base, base+1:
//   v in [0,4): base = half*8 + 2v ; v in [4,8): base = 16 + half*8 + 2(v-4)
// => fragment halfs 0..7  = P[row][kbase + half*8      .. +7]   (16B contiguous)
//    fragment halfs 8..15 = P[row][kbase + 16 + half*8 .. +7]   (16B contiguous)
__device__ __forceinline__ v16h make_frag(v8h lo, v8h hi) {
  return __builtin_shufflevector(lo, hi, 0, 1, 2, 3, 4, 5, 6, 7,
                                 8, 9, 10, 11, 12, 13, 14, 15);
}

// K-major fragment load (two ds_load_b128). Used for A tiles, and for B when
// the weight matrix is stored transposed (N-major rows, K contiguous).
__device__ __forceinline__ v16h load_frag_kmajor(const _Float16* P, int ld,
                                                 int row, int kbase, int half) {
  const _Float16* p = P + row * ld + kbase + half * 8;
  v8h lo = *(const v8h*)(p);
  v8h hi = *(const v8h*)(p + 16);
  return make_frag(lo, hi);
}

// ---------------- utility kernels ----------------
__global__ void copy4_kernel(const float4* __restrict__ in,
                             float4* __restrict__ out, int n4) {
  int i = blockIdx.x * blockDim.x + threadIdx.x;
  int stride = gridDim.x * blockDim.x;
  for (; i < n4; i += stride) out[i] = in[i];
}

// f32 [rows x cols] -> f16 transposed [cols x rows] (K-major weight layout)
__global__ void cvt_half_t_kernel(const float* __restrict__ in,
                                  _Float16* __restrict__ out, int rows, int cols) {
  int i = blockIdx.x * blockDim.x + threadIdx.x;
  if (i < rows * cols) {
    int r = i / cols, c = i % cols;
    out[c * rows + r] = (_Float16)in[i];
  }
}

// ---------------- edge scatter-add (one wave32 per edge) ----------------
__global__ void scatter_add_kernel(const float* __restrict__ feat,
                                   const int* __restrict__ src,
                                   const int* __restrict__ dst,
                                   float* __restrict__ agg, int nE) {
  const int lane = threadIdx.x & 31;
  int w = (blockIdx.x * blockDim.x + threadIdx.x) >> 5;
  const int nw = (gridDim.x * blockDim.x) >> 5;
  for (int e = w; e < nE; e += nw) {
    int s = src[e];
    int d = dst[e];
    if (e + nw < nE) {  // prefetch next gather row (global_prefetch_b8)
      __builtin_prefetch(feat + (size_t)src[e + nw] * FD, 0, 0);
    }
    float4 v = ((const float4*)(feat + (size_t)s * FD))[lane];  // 512B/wave
    float* ap = agg + (size_t)d * FD + (lane << 2);
    atomicAdd(ap + 0, v.x);
    atomicAdd(ap + 1, v.y);
    atomicAdd(ap + 2, v.z);
    atomicAdd(ap + 3, v.w);
  }
}

// ------------- layer-1 MLP: h1 = relu(relu(agg@W1a+b1a)@W1b+b1b) -------------
// one block = 32 rows x 128 cols; 8 waves; each wave owns a 16-wide col tile
// and computes two 16x16 subtiles (reusing the B fragment per K chunk).
__global__ __launch_bounds__(256) void mlp_h_kernel(
    const float* __restrict__ agg,
    const _Float16* __restrict__ wAt,  // W1a^T  [N=128][K=128]
    const _Float16* __restrict__ wBt,  // W1b^T  [N=128][K=128]
    const float* __restrict__ bA, const float* __restrict__ bB,
    float* __restrict__ h1, int n) {
  __shared__ alignas(16) _Float16 sW[FD * FD];   // 32KB, W1a^T then W1b^T
  __shared__ alignas(16) _Float16 sA[32 * FD];   // 8KB input tile (f16)
  __shared__ alignas(16) _Float16 sT[32 * FD];   // 8KB intermediate tile (f16)
  const int tid = threadIdx.x;
  const int wave = tid >> 5, lane = tid & 31;
  const int half = lane >> 4, mr = lane & 15;
  const int rowBase = blockIdx.x * 32;
  const int col = wave * 16 + mr;

  {  // stage first weight matrix (f16, 16B vectors)
    const uint4* g = (const uint4*)wAt;
    uint4* s = (uint4*)sW;
    for (int i = tid; i < FD * FD / 8; i += 256) s[i] = g[i];
  }
  // stage A tile, f32 -> f16
  for (int i = tid; i < 32 * FD / 4; i += 256) {
    int r = i >> 5;  // 32 float4 per row
    int gr = rowBase + r;
    float4 v = make_float4(0.f, 0.f, 0.f, 0.f);
    if (gr < n) v = ((const float4*)(agg + (size_t)gr * FD))[i & 31];
    int base = i * 4;
    sA[base + 0] = (_Float16)v.x;
    sA[base + 1] = (_Float16)v.y;
    sA[base + 2] = (_Float16)v.z;
    sA[base + 3] = (_Float16)v.w;
  }
  __syncthreads();

  // GEMM1: (32x128) @ (128x128) col tile
  v8f acc0 = {}, acc1 = {};
#pragma unroll
  for (int kk = 0; kk < 4; ++kk) {
    v16h b  = load_frag_kmajor(sW, FD, col, kk * 32, half);
    v16h a0 = load_frag_kmajor(sA, FD, mr,       kk * 32, half);
    v16h a1 = load_frag_kmajor(sA, FD, 16 + mr,  kk * 32, half);
    acc0 = __builtin_amdgcn_wmma_f32_16x16x32_f16(false, a0, false, b,
                                                  (short)0, acc0, false, false);
    acc1 = __builtin_amdgcn_wmma_f32_16x16x32_f16(false, a1, false, b,
                                                  (short)0, acc1, false, false);
  }
  {
    float bias = bA[col];
#pragma unroll
    for (int r = 0; r < 8; ++r) {
      float v0 = acc0[r] + bias;
      float v1 = acc1[r] + bias;
      v0 = v0 > 0.f ? v0 : 0.f;
      v1 = v1 > 0.f ? v1 : 0.f;
      int m = (half << 3) + r;
      sT[m * FD + col]        = (_Float16)v0;
      sT[(16 + m) * FD + col] = (_Float16)v1;
    }
  }
  __syncthreads();
  {  // stage second weight matrix over sW
    const uint4* g = (const uint4*)wBt;
    uint4* s = (uint4*)sW;
    for (int i = tid; i < FD * FD / 8; i += 256) s[i] = g[i];
  }
  __syncthreads();

  // GEMM2: (32x128) @ (128x128) col tile
  v8f acc2 = {}, acc3 = {};
#pragma unroll
  for (int kk = 0; kk < 4; ++kk) {
    v16h b  = load_frag_kmajor(sW, FD, col, kk * 32, half);
    v16h a0 = load_frag_kmajor(sT, FD, mr,      kk * 32, half);
    v16h a1 = load_frag_kmajor(sT, FD, 16 + mr, kk * 32, half);
    acc2 = __builtin_amdgcn_wmma_f32_16x16x32_f16(false, a0, false, b,
                                                  (short)0, acc2, false, false);
    acc3 = __builtin_amdgcn_wmma_f32_16x16x32_f16(false, a1, false, b,
                                                  (short)0, acc3, false, false);
  }
  {
    float bias = bB[col];
#pragma unroll
    for (int r = 0; r < 8; ++r) {
      float v0 = acc2[r] + bias;
      float v1 = acc3[r] + bias;
      v0 = v0 > 0.f ? v0 : 0.f;  // inter-layer ReLU fused here
      v1 = v1 > 0.f ? v1 : 0.f;
      int m = (half << 3) + r;
      int gr0 = rowBase + m;
      int gr1 = rowBase + 16 + m;
      if (gr0 < n) h1[(size_t)gr0 * FD + col] = v0;
      if (gr1 < n) h1[(size_t)gr1 * FD + col] = v1;
    }
  }
}

// ------- layer-2 MLP: out = relu(agg2@W2a+b2a)@W2b+b2b  ([N,16]) -------
// one block = 128 rows; each wave owns one 16-row tile (D_OUT = 16 cols)
__global__ __launch_bounds__(256) void mlp_out_kernel(
    const float* __restrict__ agg,
    const _Float16* __restrict__ wAt,  // W2a^T [16][128]
    const _Float16* __restrict__ wBt,  // W2b^T [16][16]
    const float* __restrict__ bA, const float* __restrict__ bB,
    float* __restrict__ out, int n) {
  __shared__ alignas(16) _Float16 sA[128 * FD];    // 32KB input rows (f16)
  __shared__ alignas(16) _Float16 sWa[16 * FD];    // 4KB  W2a^T
  __shared__ alignas(16) _Float16 sWb[16 * 16];    // 512B W2b^T
  __shared__ alignas(16) _Float16 sT[8][16 * 16];  // per-wave 16x16 intermediate
  const int tid = threadIdx.x;
  const int wave = tid >> 5, lane = tid & 31;
  const int half = lane >> 4, mr = lane & 15;
  const int rowBase = blockIdx.x * 128;

  {
    const uint4* g = (const uint4*)wAt;
    uint4* s = (uint4*)sWa;
    for (int i = tid; i < 16 * FD / 8; i += 256) s[i] = g[i];
  }
  {
    const uint4* g = (const uint4*)wBt;
    uint4* s = (uint4*)sWb;
    if (tid < 16 * 16 / 8) s[tid] = g[tid];
  }
  for (int i = tid; i < 128 * FD / 4; i += 256) {
    int r = i >> 5;
    int gr = rowBase + r;
    float4 v = make_float4(0.f, 0.f, 0.f, 0.f);
    if (gr < n) v = ((const float4*)(agg + (size_t)gr * FD))[i & 31];
    int base = i * 4;
    sA[base + 0] = (_Float16)v.x;
    sA[base + 1] = (_Float16)v.y;
    sA[base + 2] = (_Float16)v.z;
    sA[base + 3] = (_Float16)v.w;
  }
  __syncthreads();

  // GEMM1: (16x128) @ (128x16), K = 128
  v8f acc = {};
#pragma unroll
  for (int kk = 0; kk < 4; ++kk) {
    v16h a = load_frag_kmajor(sA + wave * 16 * FD, FD, mr, kk * 32, half);
    v16h b = load_frag_kmajor(sWa, FD, mr, kk * 32, half);
    acc = __builtin_amdgcn_wmma_f32_16x16x32_f16(false, a, false, b,
                                                 (short)0, acc, false, false);
  }
  {
    float bias = bA[mr];
#pragma unroll
    for (int r = 0; r < 8; ++r) {
      float v = acc[r] + bias;
      v = v > 0.f ? v : 0.f;
      sT[wave][((half << 3) + r) * 16 + mr] = (_Float16)v;  // same-wave LDS, in-order
    }
  }

  // GEMM2: (16x16) @ (16x16), K = 16 zero-padded to 32.
  // Valid K pairs occupy exactly the low 8-half group of each lane half,
  // so the fragment is one ds_load_b128 + zeroed upper group.
  v8h zero = {};
  v8h alo = *(const v8h*)(&sT[wave][mr * 16 + half * 8]);
  v8h blo = *(const v8h*)(&sWb[mr * 16 + half * 8]);
  v16h a2 = make_frag(alo, zero);
  v16h b2 = make_frag(blo, zero);
  v8f acc2 = {};
  acc2 = __builtin_amdgcn_wmma_f32_16x16x32_f16(false, a2, false, b2,
                                                (short)0, acc2, false, false);
  {
    float bias = bB[mr];
#pragma unroll
    for (int r = 0; r < 8; ++r) {
      float v = acc2[r] + bias;  // no final ReLU
      int gr = rowBase + wave * 16 + (half << 3) + r;
      if (gr < n) out[(size_t)gr * 16 + mr] = v;
    }
  }
}

extern "C" void kernel_launch(void* const* d_in, const int* in_sizes, int n_in,
                              void* d_out, int out_size, void* d_ws, size_t ws_size,
                              hipStream_t stream) {
  const float* x   = (const float*)d_in[0];
  const int*   ei  = (const int*)d_in[1];
  const float* w1a = (const float*)d_in[2];
  const float* b1a = (const float*)d_in[3];
  const float* w1b = (const float*)d_in[4];
  const float* b1b = (const float*)d_in[5];
  const float* w2a = (const float*)d_in[6];
  const float* b2a = (const float*)d_in[7];
  const float* w2b = (const float*)d_in[8];
  const float* b2b = (const float*)d_in[9];
  float* out = (float*)d_out;

  const int N = in_sizes[0] / FD;
  const int E = in_sizes[1] / 2;
  const int* src = ei;
  const int* dst = ei + E;

  // workspace layout (~51.3 MB)
  float* agg = (float*)d_ws;                            // N*128 f32 (reused per layer)
  float* h1  = agg + (size_t)N * FD;                    // N*128 f32
  _Float16* w1a_t = (_Float16*)(h1 + (size_t)N * FD);   // 128x128 (transposed)
  _Float16* w1b_t = w1a_t + FD * FD;                    // 128x128 (transposed)
  _Float16* w2a_t = w1b_t + FD * FD;                    // 16x128  (transposed)
  _Float16* w2b_t = w2a_t + FD * 16;                    // 16x16   (transposed)

  // weights f32 -> f16, transposed to K-major for ds_load_b128 B fragments
  cvt_half_t_kernel<<<(FD * FD + 255) / 256, 256, 0, stream>>>(w1a, w1a_t, FD, FD);
  cvt_half_t_kernel<<<(FD * FD + 255) / 256, 256, 0, stream>>>(w1b, w1b_t, FD, FD);
  cvt_half_t_kernel<<<(FD * 16 + 255) / 256, 256, 0, stream>>>(w2a, w2a_t, FD, 16);
  cvt_half_t_kernel<<<(16 * 16 + 255) / 256, 256, 0, stream>>>(w2b, w2b_t, 16, 16);

  const int n4 = N * FD / 4;
  const int rowTiles16 = (N + 15) / 16;
  const int rowTiles32 = (N + 31) / 32;

  // ---- layer 1 ----
  copy4_kernel<<<1024, 256, 0, stream>>>((const float4*)x, (float4*)agg, n4);
  scatter_add_kernel<<<4096, 256, 0, stream>>>(x, src, dst, agg, E);
  mlp_h_kernel<<<rowTiles32, 256, 0, stream>>>(agg, w1a_t, w1b_t, b1a, b1b, h1, N);

  // ---- layer 2 (reuse agg buffer) ----
  copy4_kernel<<<1024, 256, 0, stream>>>((const float4*)h1, (float4*)agg, n4);
  scatter_add_kernel<<<4096, 256, 0, stream>>>(h1, src, dst, agg, E);
  mlp_out_kernel<<<(rowTiles16 + 7) / 8, 256, 0, stream>>>(agg, w2a_t, w2b_t,
                                                           b2a, b2b, out, N);
}
